// Attention_14697378086932
// MI455X (gfx1250) — matrix operand order repaired and checked by reading, not compiled, not run on hardware
//
#include <hip/hip_runtime.h>
#include <hip/hip_bf16.h>

#define HIDDEN  1024
#define SEQ     2048
#define BATCHSZ 2
#define NHEADS  16
#define HEADD   64
#define MROWS   (BATCHSZ * SEQ)   // 4096
#define BSTRIDE 40                // B-panel row stride in halfs (32 data + 8 pad, 80B)
#define PANELH  (64 * BSTRIDE)    // halfs per panel buffer

typedef __attribute__((ext_vector_type(16))) _Float16 v16h;
typedef __attribute__((ext_vector_type(8)))  _Float16 v8h;
typedef __attribute__((ext_vector_type(8)))  float    v8f;
typedef __attribute__((ext_vector_type(4)))  int      v4i;

#if defined(__has_builtin)
#if __has_builtin(__builtin_amdgcn_global_load_async_to_lds_b128) && \
    __has_builtin(__builtin_amdgcn_s_wait_asynccnt)
#define HAVE_ASYNC_LDS 1
#endif
#endif

#ifdef HAVE_ASYNC_LDS
typedef __attribute__((address_space(1))) v4i gbl_v4i;   // prints as "__device__"
typedef __attribute__((address_space(3))) v4i lds_v4i;   // prints as "__shared__"
#endif

// Copy 16B global -> LDS. Async (ASYNCcnt-tracked) when the toolchain exposes
// the gfx1250 builtin; otherwise a register bounce (load_b128 + ds_store_b128).
static __device__ __forceinline__ void copy16_g2l(const _Float16* g, _Float16* l) {
#ifdef HAVE_ASYNC_LDS
    __builtin_amdgcn_global_load_async_to_lds_b128((gbl_v4i*)g, (lds_v4i*)l, 0, 0);
#else
    *(v8h*)l = *(const v8h*)g;
#endif
}

template <int N>
static __device__ __forceinline__ void wait_async_le() {
#ifdef HAVE_ASYNC_LDS
    __builtin_amdgcn_s_wait_asynccnt(N);
#endif
}

static __device__ __forceinline__ v16h combine16(v8h lo, v8h hi) {
    v16h r;
#pragma unroll
    for (int i = 0; i < 8; ++i) { r[i] = lo[i]; r[i + 8] = hi[i]; }
    return r;
}

// A operand: f16 16x32 (MxK). Source row-major, leading dim `ld` (in halfs).
// Lane L: m = L&15, kg = L>>4. halves 0..7 -> K = kg*8+h ; halves 8..15 -> 16+kg*8+h.
static __device__ __forceinline__ v16h load_a16x32(const _Float16* base, int ld, int lane) {
    const int m  = lane & 15;
    const int kg = lane >> 4;
    const _Float16* p = base + (size_t)m * ld + kg * 8;
    return combine16(*(const v8h*)(p), *(const v8h*)(p + 16));
}

// B operand: f16 32x16 (KxN). Element (k, n) at base + n*ld + k (K-contiguous).
// Lane L: n = L&15, kg = L>>4. halves 0..7 -> K = kg*8+h ; halves 8..15 -> 16+kg*8+h.
static __device__ __forceinline__ v16h load_b32x16(const _Float16* base, int ld, int lane) {
    const int n  = lane & 15;
    const int kg = lane >> 4;
    const _Float16* p = base + (size_t)n * ld + kg * 8;
    return combine16(*(const v8h*)(p), *(const v8h*)(p + 16));
}

static __device__ __forceinline__ v8f wmma16(v16h a, v16h b, v8f c) {
    // (neg_a, A, neg_b, B, c_mod, C, reuse_a, reuse_b)
    return __builtin_amdgcn_wmma_f32_16x16x32_f16(false, a, false, b, (short)0, c,
                                                  false, false);
}

// ---------------------------------------------------------------------------
__global__ void cvt_f32_f16_kernel(const float* __restrict__ in,
                                   _Float16* __restrict__ out, int n) {
    int i = blockIdx.x * blockDim.x + threadIdx.x;
    if (i < n) out[i] = (_Float16)in[i];
}

// W (dim x dim, row-major, used as x @ W) -> WT[n][k] = W[k][n] in f16
__global__ void transpose_cvt_kernel(const float* __restrict__ W,
                                     _Float16* __restrict__ WT, int dim) {
    int i = blockIdx.x * blockDim.x + threadIdx.x;
    if (i < dim * dim) {
        int nn = i / dim;
        int kk = i - nn * dim;
        WT[i] = (_Float16)W[(size_t)kk * dim + nn];
    }
}

// ---------------------------------------------------------------------------
// WMMA GEMM: C[m][n] = sum_k A[m][k] * WT[n][k] + bias[n]
// Block = 128 threads (4 waves) = 64(M) x 64(N) tile.
// B panel (64 cols x 32 k) staged in LDS, double buffered via async-to-LDS.
// mode 0: rotary on columns [0,64), store f16 row-major (Q, K)
// mode 1: store f16 transposed per (batch,col): out[(b*HIDDEN + n)*SEQ + pos] (V)
// mode 2: store f32 row-major (final output)
__global__ __launch_bounds__(128)
void gemm_wmma_kernel(const _Float16* __restrict__ A,    // MROWS x HIDDEN
                      const _Float16* __restrict__ WT,   // HIDDEN x HIDDEN (WT[n][k])
                      const float*    __restrict__ bias, // HIDDEN
                      const float*    __restrict__ rope_cos, // MROWS x 64 (mode 0)
                      const float*    __restrict__ rope_sin,
                      _Float16*       __restrict__ out_h,
                      float*          __restrict__ out_f,
                      int mode) {
    __shared__ _Float16 bpanel[2 * PANELH];

    const int lane = threadIdx.x & 31;
    const int wave = threadIdx.x >> 5;
    const int tid  = threadIdx.x;
    const int n0   = blockIdx.x * 64;               // column range of the block
    const int m0   = blockIdx.y * 64 + wave * 16;   // row tile of this wave

    const _Float16* aBase = A + (size_t)m0 * HIDDEN;

    // Cooperative B-panel copy: 256 16B segments, 2 per thread.
    const int seg0 = tid * 2, seg1 = tid * 2 + 1;
    const int col0 = seg0 >> 2, off0 = (seg0 & 3) * 8;   // halfs within 32-k row
    const int col1 = seg1 >> 2, off1 = (seg1 & 3) * 8;
    const _Float16* bsrc0 = WT + (size_t)(n0 + col0) * HIDDEN + off0;
    const _Float16* bsrc1 = WT + (size_t)(n0 + col1) * HIDDEN + off1;
    const int dst0 = col0 * BSTRIDE + off0;              // panel-relative (halfs)
    const int dst1 = col1 * BSTRIDE + off1;
    // per-wave B-operand read base (panel-relative, halfs)
    const int rdoff = (lane & 15) * BSTRIDE + (lane >> 4) * 8;

    v8f acc[4];
#pragma unroll
    for (int t = 0; t < 4; ++t) acc[t] = (v8f){};

    // prologue: chunk 0 -> buffer 0
    copy16_g2l(bsrc0, &bpanel[dst0]);
    copy16_g2l(bsrc1, &bpanel[dst1]);

#pragma unroll 1            // single body: keeps accumulators in-place (no reg rotation)
    for (int k0 = 0, i = 0; k0 < HIDDEN; k0 += 32, ++i) {
        _Float16* readP  = &bpanel[(i & 1) * PANELH];
        _Float16* writeP = &bpanel[((i & 1) ^ 1) * PANELH];
        if (k0 + 32 < HIDDEN) {
            // issue next chunk into the other buffer, then wait for the
            // oldest 2 async ops (this chunk) while next stays in flight
            copy16_g2l(bsrc0 + k0 + 32, writeP + dst0);
            copy16_g2l(bsrc1 + k0 + 32, writeP + dst1);
            wait_async_le<2>();
            __builtin_prefetch(aBase + k0 + 64, 0, 3);
        } else {
            wait_async_le<0>();
        }
        __syncthreads();   // panel for this chunk visible to all 4 waves

        const v16h av = load_a16x32(aBase + k0, HIDDEN, lane);
#pragma unroll
        for (int t = 0; t < 4; ++t) {
            const _Float16* p = readP + t * 16 * BSTRIDE + rdoff;
            const v16h bv = combine16(*(const v8h*)p, *(const v8h*)(p + 16));
            acc[t] = wmma16(av, bv, acc[t]);
        }
        __syncthreads();   // all waves done with this buffer before overwrite
    }

    // C layout: lane L holds column (L&15); VGPR j holds row m0 + (L>>4)*8 + j
    const int rbase = (lane >> 4) * 8;
#pragma unroll
    for (int t = 0; t < 4; ++t) {
        const int n = n0 + t * 16 + (lane & 15);
        const float bn = bias[n];
        float v[8];
#pragma unroll
        for (int j = 0; j < 8; ++j) v[j] = acc[t][j] + bn;

        if (mode == 0 && n0 == 0) {   // rotary covers exactly columns 0..63
#pragma unroll
            for (int j = 0; j < 8; ++j) {
                const int m = m0 + rbase + j;
                const float c = rope_cos[(size_t)m * HEADD + n];
                const float s = rope_sin[(size_t)m * HEADD + n];
                const float partner = __shfl_xor(v[j], 1, 32);   // t[n^1]
                const float rh = (n & 1) ? partner : -partner;   // rotate_half
                v[j] = v[j] * c + rh * s;
            }
        }

        if (mode == 2) {
#pragma unroll
            for (int j = 0; j < 8; ++j)
                out_f[(size_t)(m0 + rbase + j) * HIDDEN + n] = v[j];
        } else if (mode == 1) {
#pragma unroll
            for (int j = 0; j < 8; ++j) {
                const int m = m0 + rbase + j;
                const int b = m >> 11;               // SEQ = 2048
                const int pos = m & (SEQ - 1);
                out_h[((size_t)(b * HIDDEN + n)) * SEQ + pos] = (_Float16)v[j];
            }
        } else {
#pragma unroll
            for (int j = 0; j < 8; ++j)
                out_h[(size_t)(m0 + rbase + j) * HIDDEN + n] = (_Float16)v[j];
        }
    }
}

// ---------------------------------------------------------------------------
// Flash attention: one wave = 16 queries of one (batch, head); online softmax
// over 2048 keys in chunks of 32. No score scaling (matches reference).
__global__ __launch_bounds__(128)
void attn_wmma_kernel(const _Float16* __restrict__ q_h,  // (B*SEQ) x HIDDEN
                      const _Float16* __restrict__ k_h,  // (B*SEQ) x HIDDEN
                      const _Float16* __restrict__ v_t,  // (B*HIDDEN) x SEQ  (d-major)
                      _Float16*       __restrict__ attn_h) {
    __shared__ _Float16 lds_p[4][16 * 32];

    const int lane = threadIdx.x & 31;
    const int wave = threadIdx.x >> 5;
    const int bh = blockIdx.x;            // 0 .. B*NHEADS-1
    const int b  = bh >> 4;
    const int h  = bh & (NHEADS - 1);
    const int q0 = blockIdx.y * 64 + wave * 16;   // query tile base within batch

    const _Float16* qBase = q_h + ((size_t)(b * SEQ + q0)) * HIDDEN + h * HEADD;
    const _Float16* kBase = k_h + ((size_t)(b * SEQ)) * HIDDEN + h * HEADD;
    const _Float16* vBase = v_t + ((size_t)(b * HIDDEN + h * HEADD)) * SEQ;

    // Q A-operands are loop-invariant: d = 0..31 and 32..63
    const v16h aQ0 = load_a16x32(qBase +  0, HIDDEN, lane);
    const v16h aQ1 = load_a16x32(qBase + 32, HIDDEN, lane);

    float mrow[8], lrow[8];
    v8f accd[4];
#pragma unroll
    for (int j = 0; j < 8; ++j) { mrow[j] = -1e30f; lrow[j] = 0.f; }
#pragma unroll
    for (int t = 0; t < 4; ++t) accd[t] = (v8f){};

    _Float16* pw = lds_p[wave];
    const int colb  = lane & 15;
    const int rbase = (lane >> 4) * 8;

#pragma unroll 1
    for (int kb = 0; kb < SEQ; kb += 32) {
        if (kb + 32 < SEQ) {   // prefetch next K/V chunk into cache
            __builtin_prefetch(kBase + (size_t)(kb + 32) * HIDDEN, 0, 3);
            __builtin_prefetch(vBase + kb + 32, 0, 3);
        }

        // --- scores: two 16-key tiles, K-dim = head dim 64 (2 WMMAs each)
        v8f s0 = {}, s1 = {};
        {
            const _Float16* kb0 = kBase + (size_t)kb * HIDDEN;
            const _Float16* kb1 = kBase + (size_t)(kb + 16) * HIDDEN;
            s0 = wmma16(aQ0, load_b32x16(kb0 +  0, HIDDEN, lane), s0);
            s0 = wmma16(aQ1, load_b32x16(kb0 + 32, HIDDEN, lane), s0);
            s1 = wmma16(aQ0, load_b32x16(kb1 +  0, HIDDEN, lane), s1);
            s1 = wmma16(aQ1, load_b32x16(kb1 + 32, HIDDEN, lane), s1);
        }

        // --- online softmax (rows live per-VGPR j; columns across 16-lane groups)
        float p0[8], p1[8];
#pragma unroll
        for (int j = 0; j < 8; ++j) {
            float r = fmaxf(s0[j], s1[j]);
            r = fmaxf(r, __shfl_xor(r, 1, 32));
            r = fmaxf(r, __shfl_xor(r, 2, 32));
            r = fmaxf(r, __shfl_xor(r, 4, 32));
            r = fmaxf(r, __shfl_xor(r, 8, 32));
            const float mn = fmaxf(mrow[j], r);
            const float alpha = __expf(mrow[j] - mn);
            p0[j] = __expf(s0[j] - mn);
            p1[j] = __expf(s1[j] - mn);
            float rs = p0[j] + p1[j];
            rs += __shfl_xor(rs, 1, 32);
            rs += __shfl_xor(rs, 2, 32);
            rs += __shfl_xor(rs, 4, 32);
            rs += __shfl_xor(rs, 8, 32);
            lrow[j] = lrow[j] * alpha + rs;
            mrow[j] = mn;
#pragma unroll
            for (int t = 0; t < 4; ++t) accd[t][j] *= alpha;
        }

        // --- reshape probs: C layout -> A(16x32) layout via LDS
#pragma unroll
        for (int j = 0; j < 8; ++j) {
            pw[(rbase + j) * 32 + colb]      = (_Float16)p0[j];
            pw[(rbase + j) * 32 + 16 + colb] = (_Float16)p1[j];
        }
        __syncthreads();
        v16h aP;
        {
            const int m = lane & 15, kg = lane >> 4;
            const _Float16* p = pw + m * 32 + kg * 8;
            aP = combine16(*(const v8h*)(p), *(const v8h*)(p + 16));
        }
        __syncthreads();

        // --- P x V : d = 4 tiles of 16, K = 32 keys
#pragma unroll
        for (int t = 0; t < 4; ++t) {
            v16h bv = load_b32x16(vBase + (size_t)(t * 16) * SEQ + kb, SEQ, lane);
            accd[t] = wmma16(aP, bv, accd[t]);
        }
    }

    // --- epilogue: divide by row sum, store f16 row-major at head's columns
#pragma unroll
    for (int t = 0; t < 4; ++t) {
        const int d = t * 16 + (lane & 15);
#pragma unroll
        for (int j = 0; j < 8; ++j) {
            const float o = accd[t][j] / lrow[j];
            const int q = q0 + rbase + j;
            attn_h[((size_t)(b * SEQ + q)) * HIDDEN + h * HEADD + d] = (_Float16)o;
        }
    }
}

// ---------------------------------------------------------------------------
extern "C" void kernel_launch(void* const* d_in, const int* in_sizes, int n_in,
                              void* d_out, int out_size, void* d_ws, size_t ws_size,
                              hipStream_t stream) {
    const float* x        = (const float*)d_in[0];
    const float* rope_cos = (const float*)d_in[1];
    const float* rope_sin = (const float*)d_in[2];
    const float* Wq       = (const float*)d_in[3];
    const float* bq       = (const float*)d_in[4];
    const float* Wk       = (const float*)d_in[5];
    const float* bk       = (const float*)d_in[6];
    const float* Wv       = (const float*)d_in[7];
    const float* bv       = (const float*)d_in[8];
    const float* Wo       = (const float*)d_in[9];
    const float* bo       = (const float*)d_in[10];
    float* out = (float*)d_out;

    char* ws = (char*)d_ws;
    const size_t BIG = (size_t)MROWS * HIDDEN * sizeof(_Float16);   // 8 MB
    const size_t WSZ = (size_t)HIDDEN * HIDDEN * sizeof(_Float16);  // 2 MB
    _Float16* x_h    = (_Float16*)ws;             ws += BIG;
    _Float16* q_h    = (_Float16*)ws;             ws += BIG;
    _Float16* k_h    = (_Float16*)ws;             ws += BIG;
    _Float16* v_t    = (_Float16*)ws;             ws += BIG;   // (B*HIDDEN) x SEQ
    _Float16* attn_h = (_Float16*)ws;             ws += BIG;
    _Float16* wq_t   = (_Float16*)ws;             ws += WSZ;
    _Float16* wk_t   = (_Float16*)ws;             ws += WSZ;
    _Float16* wv_t   = (_Float16*)ws;             ws += WSZ;
    _Float16* wo_t   = (_Float16*)ws;             ws += WSZ;

    const int nx = MROWS * HIDDEN;
    cvt_f32_f16_kernel<<<(nx + 255) / 256, 256, 0, stream>>>(x, x_h, nx);
    const int nw = HIDDEN * HIDDEN;
    transpose_cvt_kernel<<<(nw + 255) / 256, 256, 0, stream>>>(Wq, wq_t, HIDDEN);
    transpose_cvt_kernel<<<(nw + 255) / 256, 256, 0, stream>>>(Wk, wk_t, HIDDEN);
    transpose_cvt_kernel<<<(nw + 255) / 256, 256, 0, stream>>>(Wv, wv_t, HIDDEN);
    transpose_cvt_kernel<<<(nw + 255) / 256, 256, 0, stream>>>(Wo, wo_t, HIDDEN);

    dim3 gg(HIDDEN / 64, MROWS / 64);   // 16 x 64 blocks, 128 threads (4 waves)
    gemm_wmma_kernel<<<gg, 128, 0, stream>>>(x_h, wq_t, bq, rope_cos, rope_sin,
                                             q_h, nullptr, 0);
    gemm_wmma_kernel<<<gg, 128, 0, stream>>>(x_h, wk_t, bk, rope_cos, rope_sin,
                                             k_h, nullptr, 0);
    gemm_wmma_kernel<<<gg, 128, 0, stream>>>(x_h, wv_t, bv, nullptr, nullptr,
                                             v_t, nullptr, 1);

    attn_wmma_kernel<<<dim3(BATCHSZ * NHEADS, SEQ / 64), 128, 0, stream>>>(
        q_h, k_h, v_t, attn_h);

    gemm_wmma_kernel<<<gg, 128, 0, stream>>>(attn_h, wo_t, bo, nullptr, nullptr,
                                             nullptr, out, 2);
    (void)in_sizes; (void)n_in; (void)out_size; (void)ws_size;
}
